// Set2Set_77635828843229
// MI455X (gfx1250) — compile-verified, matrix operand order checked by reading, not run on hardware
//
#include <hip/hip_runtime.h>
#include <cstddef>
#include <cstdint>

typedef __attribute__((ext_vector_type(16))) _Float16 v16h;
typedef __attribute__((ext_vector_type(8)))  _Float16 v8h;
typedef __attribute__((ext_vector_type(8)))  float    v8f;

// ---------------------------------------------------------------------------
// Helpers
// ---------------------------------------------------------------------------
__device__ __forceinline__ void atomicMaxFloat(float* addr, float v) {
    // Works with init = -inf: non-negative floats order like ints,
    // negative floats order inversely like unsigned ints.
    if (v >= 0.0f) atomicMax((int*)addr, __float_as_int(v));
    else           atomicMin((unsigned int*)addr, (unsigned int)__float_as_int(v));
}

__device__ __forceinline__ float sigmoidf_(float x) {
    return 1.0f / (1.0f + __expf(-x));
}

// ---------------------------------------------------------------------------
// One-time prep: Wc16[512][256] = fp16(w_ih + [w_hh | 0]), bias = b_ih + b_hh
// ---------------------------------------------------------------------------
__global__ void prep_kernel(const float* __restrict__ w_ih,
                            const float* __restrict__ w_hh,
                            const float* __restrict__ b_ih,
                            const float* __restrict__ b_hh,
                            _Float16* __restrict__ Wc16,
                            float* __restrict__ bias) {
    const int n = blockIdx.x;     // 0..511
    const int k = threadIdx.x;    // 0..255
    float w = w_ih[n * 256 + k];
    if (k < 128) w += w_hh[n * 128 + k];
    Wc16[n * 256 + k] = (_Float16)w;
    if (k == 0) bias[n] = b_ih[n] + b_hh[n];
}

__global__ void zero_state_kernel(float* __restrict__ h, float* __restrict__ c,
                                  float* __restrict__ r, int total) {
    int i = blockIdx.x * blockDim.x + threadIdx.x;
    if (i < total) { h[i] = 0.0f; c[i] = 0.0f; r[i] = 0.0f; }
}

__global__ void seg_init_kernel(float* __restrict__ r, float* __restrict__ emax,
                                float* __restrict__ denom, int B) {
    int i = blockIdx.x * blockDim.x + threadIdx.x;
    if (i < B * 128) r[i] = 0.0f;
    if (i < B) { emax[i] = -__builtin_huge_valf(); denom[i] = 0.0f; }
}

// ---------------------------------------------------------------------------
// Fused GEMM + LSTM cell.
//   gates[B,512] = [h | readout][B,256] @ Wc16^T  + bias
// Block: 512 threads (16 waves). M-tile = 64 rows, N = full 512.
// Wave (waveM, waveN): rows [tile+16*waveM, +16), d-columns [32*waveN, +32).
// Each wave holds accumulators for all 4 gates of its d-slice, so the
// i/f/g/o activation + cell update happens entirely in registers.
// B tile (f16 weights) is staged with ASYNCcnt-tracked DMA
// (global_load_async_to_lds_b128), overlapping the A-tile f32->f16 staging.
// ---------------------------------------------------------------------------
__global__ __launch_bounds__(512, 2)
void gemm_lstm_kernel(float* __restrict__ h_state,          // [B,128] in/out
                      const float* __restrict__ r_state,    // [B,128]
                      const _Float16* __restrict__ Wc16,    // [512,256]
                      const float* __restrict__ bias,       // [512]
                      float* __restrict__ c_state,          // [B,128] in/out
                      int B) {
    __shared__ _Float16 As[64][40];    // 64 rows x 32 K (pad->40, 16B aligned rows)
    __shared__ _Float16 Bs[512][40];   // 512 N-rows x 32 K
    __shared__ float    bsh[512];

    const int tid  = threadIdx.x;
    const int tile = blockIdx.x * 64;
    const int lane = tid & 31;
    const int wid  = tid >> 5;           // 0..15
    const int waveM = wid >> 2;          // 0..3
    const int waveN = wid & 3;           // 0..3

    bsh[tid] = bias[tid];

    v8f acc[4][2] = {};                  // [gate i/f/g/o][d-frag 0/1]

    const int arowLoad = tid >> 3;       // 0..63 (A-tile staging row)
    const int akk      = (tid & 7) * 4;  // K offset within tile
    const int growLoad = tile + arowLoad;
    const bool ainb    = growLoad < B;

    // Per-thread LDS destination (low 32 bits of generic address == LDS addr)
    const unsigned ldsB = (unsigned)(uintptr_t)&Bs[tid][0];

    for (int k0 = 0; k0 < 256; k0 += 32) {
        // ---- B tile via async DMA to LDS: 64B weight row slice per thread.
        // INST_OFFSET is applied to both global and LDS addresses.
        {
            const unsigned long long ga =
                (unsigned long long)(uintptr_t)(Wc16 + (size_t)tid * 256 + k0);
            asm volatile(
                "global_load_async_to_lds_b128 %0, %1, off\n\t"
                "global_load_async_to_lds_b128 %0, %1, off offset:16\n\t"
                "global_load_async_to_lds_b128 %0, %1, off offset:32\n\t"
                "global_load_async_to_lds_b128 %0, %1, off offset:48"
                :: "v"(ldsB), "v"(ga) : "memory");
        }
        // ---- stage A tile (fp32 -> f16): A = [h | readout] rows of this block
        {
            const int kg = k0 + akk;
            float4 v = make_float4(0.f, 0.f, 0.f, 0.f);
            if (ainb) {
                const float* src = (kg < 128)
                    ? (h_state + (size_t)growLoad * 128 + kg)
                    : (r_state + (size_t)growLoad * 128 + (kg - 128));
                v = *(const float4*)src;
            }
            As[arowLoad][akk + 0] = (_Float16)v.x;
            As[arowLoad][akk + 1] = (_Float16)v.y;
            As[arowLoad][akk + 2] = (_Float16)v.z;
            As[arowLoad][akk + 3] = (_Float16)v.w;
        }
        // drain the async DMA before publishing the tile to the workgroup
        asm volatile("s_wait_asynccnt 0x0" ::: "memory");
        __syncthreads();

        // ---- A fragment (16x32 f16): lanes 0-15 hold K{0..7,16..23},
        //      lanes 16-31 hold K{8..15,24..31} (ISA 16-bit A layout)
        const int kbase = (lane < 16) ? 0 : 8;
        v16h a;
        {
            v8h* ap = (v8h*)&a;
            const int ar = waveM * 16 + (lane & 15);
            ap[0] = *(const v8h*)&As[ar][kbase];
            ap[1] = *(const v8h*)&As[ar][kbase + 16];
        }
        // ---- 8 fragments: 4 gates x 2 d-subtiles
        #pragma unroll
        for (int g = 0; g < 4; ++g) {
            #pragma unroll
            for (int j = 0; j < 2; ++j) {
                const int ncol = g * 128 + waveN * 32 + j * 16 + (lane & 15);
                v16h bf;
                v8h* bp = (v8h*)&bf;
                bp[0] = *(const v8h*)&Bs[ncol][kbase];
                bp[1] = *(const v8h*)&Bs[ncol][kbase + 16];
                acc[g][j] = __builtin_amdgcn_wmma_f32_16x16x32_f16(
                    false, a, false, bf, (short)0, acc[g][j], false, false);
            }
        }
        __syncthreads();
    }

    // ---- epilogue: LSTM cell update, all in registers.
    // D layout: lane 0-15 -> N=lane, VGPR v -> M=v (+8 for lanes 16-31).
    const int rbase = tile + waveM * 16 + ((lane >= 16) ? 8 : 0);
    const int d0    = waveN * 32 + (lane & 15);
    #pragma unroll
    for (int j = 0; j < 2; ++j) {
        const int d = d0 + j * 16;
        const float bi = bsh[d];
        const float bf_ = bsh[128 + d];
        const float bg = bsh[256 + d];
        const float bo = bsh[384 + d];
        #pragma unroll
        for (int v = 0; v < 8; ++v) {
            const int row = rbase + v;
            if (row < B) {
                const float iv = sigmoidf_(acc[0][j][v] + bi);
                const float fv = sigmoidf_(acc[1][j][v] + bf_);
                const float gv = tanhf(acc[2][j][v] + bg);
                const float ov = sigmoidf_(acc[3][j][v] + bo);
                const size_t idx = (size_t)row * 128 + d;
                const float cn = fv * c_state[idx] + iv * gv;
                c_state[idx] = cn;
                h_state[idx] = ov * tanhf(cn);
            }
        }
    }
}

// ---------------------------------------------------------------------------
// Attention pass 1: e[n] = dot(x[n], h[batch[n]]); segment max via atomics.
// One wave per node; lane holds float4 (128 floats / 32 lanes).
// ---------------------------------------------------------------------------
__global__ __launch_bounds__(256)
void attn_e_kernel(const float* __restrict__ x, const long long* __restrict__ batch,
                   const float* __restrict__ h, float* __restrict__ e,
                   float* __restrict__ emax, int N) {
    const int n    = (blockIdx.x * blockDim.x + threadIdx.x) >> 5;
    const int lane = threadIdx.x & 31;
    if (n >= N) return;
    const long long b = batch[n];
    __builtin_prefetch(x + (size_t)(n + 8) * 128, 0, 0);
    const float4 xv = ((const float4*)(x + (size_t)n * 128))[lane];
    const float4 hv = ((const float4*)(h + (size_t)b * 128))[lane];
    float p = xv.x * hv.x + xv.y * hv.y + xv.z * hv.z + xv.w * hv.w;
    #pragma unroll
    for (int off = 16; off > 0; off >>= 1) p += __shfl_xor(p, off);
    if (lane == 0) {
        e[n] = p;
        atomicMaxFloat(&emax[b], p);
    }
}

// ---------------------------------------------------------------------------
// Attention pass 2: ex = exp(e - emax); denom += ex  (e overwritten with ex)
// ---------------------------------------------------------------------------
__global__ __launch_bounds__(256)
void attn_exp_kernel(const long long* __restrict__ batch, float* __restrict__ e,
                     const float* __restrict__ emax, float* __restrict__ denom, int N) {
    const int n = blockIdx.x * blockDim.x + threadIdx.x;
    if (n >= N) return;
    const long long b = batch[n];
    const float ex = __expf(e[n] - emax[b]);
    e[n] = ex;
    atomicAdd(&denom[b], ex);
}

// ---------------------------------------------------------------------------
// Attention pass 3: readout[b] += alpha * x[n].  batch is sorted, so each wave
// walks 16 consecutive nodes and accumulates per-segment runs in registers,
// flushing with atomics only on segment boundaries (~10x fewer atomics).
// ---------------------------------------------------------------------------
__global__ __launch_bounds__(256)
void attn_readout_kernel(const float* __restrict__ x, const long long* __restrict__ batch,
                         const float* __restrict__ e, const float* __restrict__ denom,
                         float* __restrict__ readout, int N) {
    const int wave = (blockIdx.x * blockDim.x + threadIdx.x) >> 5;
    const int lane = threadIdx.x & 31;
    const int n0 = wave * 16;
    if (n0 >= N) return;
    const int n1 = min(n0 + 16, N);

    long long cur = -1;
    float scale = 0.0f;
    float4 acc = make_float4(0.f, 0.f, 0.f, 0.f);

    for (int n = n0; n < n1; ++n) {
        const long long b = batch[n];
        if (b != cur) {
            if (cur >= 0) {
                float* dst = readout + (size_t)cur * 128 + lane * 4;
                atomicAdd(dst + 0, acc.x);
                atomicAdd(dst + 1, acc.y);
                atomicAdd(dst + 2, acc.z);
                atomicAdd(dst + 3, acc.w);
            }
            cur = b;
            scale = 1.0f / denom[b];
            acc = make_float4(0.f, 0.f, 0.f, 0.f);
        }
        __builtin_prefetch(x + (size_t)(n + 4) * 128, 0, 0);
        const float w = e[n] * scale;
        const float4 xv = ((const float4*)(x + (size_t)n * 128))[lane];
        acc.x += w * xv.x; acc.y += w * xv.y; acc.z += w * xv.z; acc.w += w * xv.w;
    }
    if (cur >= 0) {
        float* dst = readout + (size_t)cur * 128 + lane * 4;
        atomicAdd(dst + 0, acc.x);
        atomicAdd(dst + 1, acc.y);
        atomicAdd(dst + 2, acc.z);
        atomicAdd(dst + 3, acc.w);
    }
}

// ---------------------------------------------------------------------------
// Final: q_star = [h | readout]
// ---------------------------------------------------------------------------
__global__ void pack_kernel(const float* __restrict__ h, const float* __restrict__ r,
                            float* __restrict__ out, int B) {
    const int i = blockIdx.x * blockDim.x + threadIdx.x;
    if (i < B * 128) {
        const int b = i >> 7, d = i & 127;
        out[(size_t)b * 256 + d]       = h[i];
        out[(size_t)b * 256 + 128 + d] = r[i];
    }
}

// ---------------------------------------------------------------------------
extern "C" void kernel_launch(void* const* d_in, const int* in_sizes, int n_in,
                              void* d_out, int out_size, void* d_ws, size_t ws_size,
                              hipStream_t stream) {
    const float*     x     = (const float*)d_in[0];
    const long long* batch = (const long long*)d_in[1];
    const float*     w_ih  = (const float*)d_in[2];
    const float*     w_hh  = (const float*)d_in[3];
    const float*     b_ih  = (const float*)d_in[4];
    const float*     b_hh  = (const float*)d_in[5];

    const int N = in_sizes[0] / 128;   // nodes
    const int B = out_size / 256;      // graphs (q_star is [B, 256])

    // workspace carve-out (256B aligned)
    char* p = (char*)d_ws;
    auto carve = [&p](size_t bytes) -> char* {
        char* r = p;
        p += (bytes + 255) & ~(size_t)255;
        return r;
    };
    _Float16* Wc16 = (_Float16*)carve((size_t)512 * 256 * sizeof(_Float16));
    float* bias    = (float*)carve(512 * sizeof(float));
    float* h32     = (float*)carve((size_t)B * 128 * sizeof(float));
    float* c32     = (float*)carve((size_t)B * 128 * sizeof(float));
    float* r32     = (float*)carve((size_t)B * 128 * sizeof(float));
    float* ebuf    = (float*)carve((size_t)N * sizeof(float));
    float* emax    = (float*)carve((size_t)B * sizeof(float));
    float* denom   = (float*)carve((size_t)B * sizeof(float));

    const int stBlocks = (B * 128 + 255) / 256;

    prep_kernel<<<512, 256, 0, stream>>>(w_ih, w_hh, b_ih, b_hh, Wc16, bias);
    zero_state_kernel<<<stBlocks, 256, 0, stream>>>(h32, c32, r32, B * 128);

    for (int it = 0; it < 6; ++it) {
        gemm_lstm_kernel<<<(B + 63) / 64, 512, 0, stream>>>(h32, r32, Wc16, bias, c32, B);
        seg_init_kernel<<<stBlocks, 256, 0, stream>>>(r32, emax, denom, B);
        attn_e_kernel<<<(N + 7) / 8, 256, 0, stream>>>(x, batch, h32, ebuf, emax, N);
        attn_exp_kernel<<<(N + 255) / 256, 256, 0, stream>>>(batch, ebuf, emax, denom, N);
        attn_readout_kernel<<<(((N + 15) / 16) + 7) / 8, 256, 0, stream>>>(
            x, batch, ebuf, denom, r32, N);
    }
    pack_kernel<<<stBlocks, 256, 0, stream>>>(h32, r32, (float*)d_out, B);
}
